// GatingNetwork_88158498718385
// MI455X (gfx1250) — matrix operand order repaired and checked by reading, not compiled, not run on hardware
//
#include <hip/hip_runtime.h>
#include <hip/hip_bf16.h>
#include <math.h>

// ---------------------------------------------------------------------------
// GatingNetwork: logits = -||w_e - x_t||, top-2 softmax scattered to (B, E)
//   B=4096 tokens, D=1024 (x = concat -> 2048), E=64 experts.
//
// ||w-x||^2 = ||w||^2 - 2 w.x + ||x||^2. The w.x GEMM runs on the bf16 WMMA
// pipe with a truncation hi/lo split (3x v_wmma_f32_16x16x32_bf16 per k-step)
// for ~fp32 accuracy. W is pre-split into bf16 hi/lo planes in d_ws
// (L2-resident, 0.5 MB). ||x||^2 is accumulated in-loop (packed, multi-chain).
// K is split in half across wave pairs (half boundary == t1/t2 concat
// boundary -> one constant A base pointer per wave); partials combine via
// LDS. 256 WGs x 8 waves = 2048 waves. Two WMMA accumulator chains break the
// per-step RAW chain. HBM traffic ~36 MB -> ~1.55 us floor at 23.3 TB/s.
// ---------------------------------------------------------------------------

typedef __attribute__((ext_vector_type(16))) __bf16 v16bf;
typedef __attribute__((ext_vector_type(8)))  float  v8f;

union BF16x16 {
    v16bf v;
    unsigned short u[16];
    unsigned int   d[8];
};

// One v_perm_b32: pack high halves of (v0, v1) -> {bf16(v1), bf16(v0)}
__device__ __forceinline__ unsigned int pack_trunc_bf16(float v0, float v1) {
    return __builtin_amdgcn_perm(__float_as_uint(v1), __float_as_uint(v0),
                                 0x07060302u);
}
__device__ __forceinline__ float trunc_bf16_f(float v) {
    return __uint_as_float(__float_as_uint(v) & 0xffff0000u);
}

// ---------------------------------------------------------------------------
// Pass 1: split gate_weight (64 x 2048) into bf16 hi/lo planes + ||w_e||^2.
// One wave32 per expert row; 8 blocks x 256 threads.
// ---------------------------------------------------------------------------
__global__ __launch_bounds__(256)
void gating_prep_w_kernel(const float* __restrict__ gw,
                          unsigned short* __restrict__ whi,
                          unsigned short* __restrict__ wlo,
                          float* __restrict__ nw) {
    const int row  = blockIdx.x * 8 + (threadIdx.x >> 5);   // 0..63
    const int lane = threadIdx.x & 31;
    const float* p = gw + (size_t)row * 2048;
    unsigned int* ph = (unsigned int*)(whi + (size_t)row * 2048);
    unsigned int* pl = (unsigned int*)(wlo + (size_t)row * 2048);

    float s = 0.0f;
    for (int c = lane; c < 1024; c += 32) {           // c indexes float PAIRS
        float v0 = p[2 * c], v1 = p[2 * c + 1];
        s = fmaf(v0, v0, s);
        s = fmaf(v1, v1, s);
        float r0 = v0 - trunc_bf16_f(v0);
        float r1 = v1 - trunc_bf16_f(v1);
        ph[c] = pack_trunc_bf16(v0, v1);
        pl[c] = pack_trunc_bf16(r0, r1);
    }
    #pragma unroll
    for (int off = 16; off > 0; off >>= 1) s += __shfl_xor(s, off, 32);
    if (lane == 0) nw[row] = s;
}

// ---------------------------------------------------------------------------
// Pass 2: S = X . W^T via WMMA with split-K(2); fused ||x||^2 and
// logit = -sqrt(||w||^2 - 2S + ||x||^2) epilogue.
// 8 waves/WG: wave = (kh<<2)|nt ; nt = expert tile 0..3, kh = K half 0/1.
// grid = 4096/16 = 256 workgroups of 256 threads.
// ---------------------------------------------------------------------------
__global__ __launch_bounds__(256)
void gating_gemm_kernel(const float* __restrict__ t1,
                        const float* __restrict__ t2,
                        const unsigned short* __restrict__ whi,
                        const unsigned short* __restrict__ wlo,
                        const float* __restrict__ nw_arr,
                        float* __restrict__ logits) {
    __shared__ float accbuf[4][32][8];   // kh=1 partial accumulators
    __shared__ float sqbuf[4][32];       // kh=1 partial ||x||^2

    const int lane  = threadIdx.x & 31;
    const int w     = threadIdx.x >> 5;  // 0..7
    const int nt    = w & 3;             // expert tile
    const int kh    = w >> 2;            // K half: 0 -> t1 cols, 1 -> t2 cols
    const int g     = lane >> 4;         // half-wave selector
    const int l16   = lane & 15;
    const int tileM = blockIdx.x * 16;

    // A layout (16x32 bf16): lane holds row M=l16; 16 contiguous K
    // [16g..16g+15] in elements 0..15.
    const int rowA   = tileM + l16;
    // B layout (32x16 bf16): lane holds column N=l16 (expert), 16 contiguous
    // K starting at 16*g.
    const int expert = nt * 16 + l16;

    // K half [1024*kh, 1024*kh+1024) lies entirely in t1 (kh=0) or t2 (kh=1).
    const float* abase = (kh == 0 ? t1 : t2) + (size_t)rowA * 1024;
    const unsigned short* bh = whi + (size_t)expert * 2048 + kh * 1024;
    const unsigned short* bl = wlo + (size_t)expert * 2048 + kh * 1024;

    v8f acc0 = {};           // hi*hi chain
    v8f acc1 = {};           // lo*hi + hi*lo chain
    float2 sqa = {0.f, 0.f}; // two packed ||x||^2 chains (4 scalar lanes)
    float2 sqb = {0.f, 0.f};

    for (int k = 0; k < 1024; k += 32) {
        // ---- A: two 8-float chunks from the single base pointer
        float av[16];
        {
            const float4* s1 = (const float4*)(abase + k + 16 * g);
            const float4* s2 = (const float4*)(abase + k + 16 * g + 8);
            float4 p0 = s1[0], p1 = s1[1], p2 = s2[0], p3 = s2[1];
            av[0]=p0.x;  av[1]=p0.y;  av[2]=p0.z;  av[3]=p0.w;
            av[4]=p1.x;  av[5]=p1.y;  av[6]=p1.z;  av[7]=p1.w;
            av[8]=p2.x;  av[9]=p2.y;  av[10]=p2.z; av[11]=p2.w;
            av[12]=p3.x; av[13]=p3.y; av[14]=p3.z; av[15]=p3.w;
        }
        // ---- ||x||^2 partials (packed fma, 2 independent chains)
        //      + truncation hi/lo split (perm-packed)
        BF16x16 ahi, alo;
        #pragma unroll
        for (int p = 0; p < 8; ++p) {
            float v0 = av[2 * p], v1 = av[2 * p + 1];
            if (p & 1) {
                sqb.x = fmaf(v0, v0, sqb.x);
                sqb.y = fmaf(v1, v1, sqb.y);
            } else {
                sqa.x = fmaf(v0, v0, sqa.x);
                sqa.y = fmaf(v1, v1, sqa.y);
            }
            ahi.d[p] = pack_trunc_bf16(v0, v1);
            float r0 = v0 - trunc_bf16_f(v0);
            float r1 = v1 - trunc_bf16_f(v1);
            alo.d[p] = pack_trunc_bf16(r0, r1);
        }
        // ---- B: pre-split bf16, 16 contiguous values (32B) per plane
        BF16x16 bhi, blo;
        {
            const int kb = k + 16 * g;
            uint4 q0 = ((const uint4*)(bh + kb))[0];
            uint4 q1 = ((const uint4*)(bh + kb))[1];
            uint4 r0 = ((const uint4*)(bl + kb))[0];
            uint4 r1 = ((const uint4*)(bl + kb))[1];
            bhi.d[0]=q0.x; bhi.d[1]=q0.y; bhi.d[2]=q0.z; bhi.d[3]=q0.w;
            bhi.d[4]=q1.x; bhi.d[5]=q1.y; bhi.d[6]=q1.z; bhi.d[7]=q1.w;
            blo.d[0]=r0.x; blo.d[1]=r0.y; blo.d[2]=r0.z; blo.d[3]=r0.w;
            blo.d[4]=r1.x; blo.d[5]=r1.y; blo.d[6]=r1.z; blo.d[7]=r1.w;
        }
        // ---- hi*hi -> acc0 ; lo*hi + hi*lo -> acc1 (two RAW chains)
        acc0 = __builtin_amdgcn_wmma_f32_16x16x32_bf16(false, ahi.v, false, bhi.v,
                                                       (short)0, acc0, false, false);
        acc1 = __builtin_amdgcn_wmma_f32_16x16x32_bf16(false, alo.v, false, bhi.v,
                                                       (short)0, acc1, false, false);
        acc1 = __builtin_amdgcn_wmma_f32_16x16x32_bf16(false, ahi.v, false, blo.v,
                                                       (short)0, acc1, false, false);
    }

    float sq = (sqa.x + sqa.y) + (sqb.x + sqb.y);

    // ---- split-K combine through LDS
    if (kh == 1) {
        #pragma unroll
        for (int r = 0; r < 8; ++r) accbuf[nt][lane][r] = acc0[r] + acc1[r];
        sqbuf[nt][lane] = sq;
    }
    __syncthreads();
    if (kh == 0) {
        sq += sqbuf[nt][lane];
        // lanes (l16) and (l16+16) jointly cover the full K range of row l16
        sq += __shfl_xor(sq, 16, 32);   // lanes m, m+16 hold nx[tileM+m]

        const float nw = nw_arr[expert];
        // C/D layout -> element r on lane L is (M = r + 8*(L/16), N = L%16)
        #pragma unroll
        for (int r = 0; r < 8; ++r) {
            const float dot = (acc0[r] + acc1[r]) + accbuf[nt][lane][r];
            const int   m   = r + 8 * g;
            const float nx  = __shfl(sq, m, 32);    // broadcast row-m norm
            const int   tok = tileM + m;
            float d2 = nw + nx - 2.0f * dot;
            d2 = d2 > 0.0f ? d2 : 0.0f;
            logits[(size_t)tok * 64 + expert] = -__builtin_sqrtf(d2);
        }
    }
}

// ---------------------------------------------------------------------------
// Pass 3: in-place top-2 + softmax over each token's 64 logits.
// One wave32 per token; butterfly top-2 merge with lower-index tie-break.
// ---------------------------------------------------------------------------
__global__ __launch_bounds__(256)
void gating_topk_kernel(float* __restrict__ out) {
    const int t    = blockIdx.x * 8 + (threadIdx.x >> 5);
    const int lane = threadIdx.x & 31;
    float* row = out + (size_t)t * 64;

    float v0 = row[lane];
    float v1 = row[lane + 32];

    float b, s; int bi, si;
    if (v0 >= v1) { b = v0; bi = lane;      s = v1; si = lane + 32; }
    else          { b = v1; bi = lane + 32; s = v0; si = lane;      }

    #pragma unroll
    for (int off = 16; off > 0; off >>= 1) {
        float ob = __shfl_xor(b,  off, 32);
        int  obi = __shfl_xor(bi, off, 32);
        float os = __shfl_xor(s,  off, 32);
        int  osi = __shfl_xor(si, off, 32);
        if (ob > b || (ob == b && obi < bi)) {
            if (b > os || (b == os && bi < osi)) { s = b; si = bi; }
            else                                 { s = os; si = osi; }
            b = ob; bi = obi;
        } else if (ob > s || (ob == s && obi < si)) {
            s = ob; si = obi;
        }
    }

    const float e2 = __expf(s - b);
    const float w1 = 1.0f / (1.0f + e2);
    const float w2 = e2 * w1;

    row[lane]      = (bi == lane)      ? w1 : (si == lane)      ? w2 : 0.0f;
    row[lane + 32] = (bi == lane + 32) ? w1 : (si == lane + 32) ? w2 : 0.0f;
}

// ---------------------------------------------------------------------------
extern "C" void kernel_launch(void* const* d_in, const int* in_sizes, int n_in,
                              void* d_out, int out_size, void* d_ws, size_t ws_size,
                              hipStream_t stream) {
    const float* t1 = (const float*)d_in[0];   // (4096, 1024)
    const float* t2 = (const float*)d_in[1];   // (4096, 1024)
    const float* gw = (const float*)d_in[2];   // (64, 2048)
    float* out = (float*)d_out;                // (4096, 64)

    // d_ws layout: whi[64*2048] bf16 | wlo[64*2048] bf16 | nw[64] f32
    unsigned short* whi = (unsigned short*)d_ws;
    unsigned short* wlo = whi + 64 * 2048;
    float*          nw  = (float*)(wlo + 64 * 2048);

    gating_prep_w_kernel<<<  8, 256, 0, stream>>>(gw, whi, wlo, nw);
    gating_gemm_kernel  <<<256, 256, 0, stream>>>(t1, t2, whi, wlo, nw, out);
    gating_topk_kernel  <<<512, 256, 0, stream>>>(out);
}
